// MSDeformableAttention3D_15006615734129
// MI455X (gfx1250) — compile-verified, matrix operand order checked by reading, not run on hardware
//
#include <hip/hip_runtime.h>
#include <hip/hip_bf16.h>
#include <stdint.h>

typedef __attribute__((ext_vector_type(16))) _Float16 v16h;
typedef __attribute__((ext_vector_type(8)))  float    v8f;
typedef __attribute__((ext_vector_type(4)))  uint32_t u32x4;
typedef __attribute__((ext_vector_type(8)))  int32_t  i32x8;
typedef __attribute__((ext_vector_type(4)))  int32_t  i32x4;

#define EMBED  256
#define HEADS  8
#define DHEAD  32
#define NPTS   8

#if defined(__has_builtin)
#if __has_builtin(__builtin_amdgcn_tensor_load_to_lds)
#define HAS_TDM 1
#endif
#endif
#ifndef HAS_TDM
#define HAS_TDM 0
#endif

// ---------------------------------------------------------------------------
// Pack a KxN (K=256) row-major f32 weight matrix into split-f16 WMMA
// B-fragment order. For each fragment slot we store 16 hi halves followed by
// 16 lo halves (lo = f32 - (f32)hi), so bh/bl are adjacent 32B vectors.
//   t = (((nt*8 + ks)*32 + lane)*2 + sel)*16 + e
//   k   = ks*32 + (e&7) + ((e>>3)<<4) + ((lane>>4)<<3)
//   col = nt*16 + (lane&15)
// (16-bit B VGPR layout per ISA 7.12.2; B mirrors A with lane indexing N.)
// ---------------------------------------------------------------------------
__global__ void pack_b_kernel(const float* __restrict__ B,
                              _Float16* __restrict__ P, int N) {
    int t = blockIdx.x * 256 + threadIdx.x;
    if (t >= EMBED * N * 2) return;
    int e2   = t & 31;
    int lane = (t >> 5) & 31;
    int ks   = (t >> 10) & 7;
    int nt   = t >> 13;
    int e    = e2 & 15;
    int sel  = e2 >> 4;          // 0 = hi, 1 = lo
    int kg   = lane >> 4;
    int k    = ks * 32 + (e & 7) + ((e >> 3) << 4) + (kg << 3);
    int col  = nt * 16 + (lane & 15);
    float v  = B[k * N + col];
    _Float16 hi = (_Float16)v;
    P[t] = sel ? (_Float16)(v - (float)hi) : hi;
}

// ---------------------------------------------------------------------------
// C[M,N] = A[M,256] * B[256,N] + bias[N]  with 3-term split-f16 WMMA
// (error ~2^-22, near-fp32). A tile (16x256 f32) staged to LDS by the Tensor
// Data Mover with pad_enable giving a 260-float row stride (bank spreading);
// fragments are split hi/lo in-register. One wave per 16x16 tile.
// ---------------------------------------------------------------------------
__global__ __launch_bounds__(256)
void gemm_wmma_kernel(const float* __restrict__ A,
                      const _Float16* __restrict__ Bp,
                      const float* __restrict__ bias,
                      float* __restrict__ C, int N) {
    __shared__ float Asf[16 * 260];   // 16 rows x 256 f32, +4 dword pad / row

    const int m0  = blockIdx.x * 16;
    const int tid = threadIdx.x;

#if HAS_TDM
    if (tid == 0) {
        uint64_t ga  = (uint64_t)(uintptr_t)(A + (size_t)m0 * EMBED);
        uint32_t lds = (uint32_t)(uintptr_t)(&Asf[0]);
        u32x4 g0;
        i32x8 g1;
        i32x4 g2 = {0, 0, 0, 0}, g3 = {0, 0, 0, 0};
        // D# group0: count=1, lds_addr, global_addr[56:0], type=2 (image)
        g0[0] = 1u;
        g0[1] = lds;
        g0[2] = (uint32_t)ga;
        g0[3] = (uint32_t)((ga >> 32) & 0x1FFFFFFull) | (2u << 30);
        // D# group1: data_size=4B(2), pad_enable, pad_interval=256dw(7),
        // pad_amount=4dw(3); tensor_dim0=256, tensor_dim1=16,
        // tile_dim0=256, tile_dim1=16, tensor_dim0_stride=256
        g1[0] = (int)((2u << 16) | (1u << 20) | (7u << 22) | (3u << 25));
        g1[1] = (int)(256u << 16);   // tensor_dim0[15:0] @ bits 63:48
        g1[2] = (int)(16u << 16);    // tensor_dim1[15:0] @ bits 95:80
        g1[3] = (int)(256u << 16);   // tile_dim0 @ bits 127:112
        g1[4] = 16;                  // tile_dim1 @ bits 143:128
        g1[5] = 256;                 // tensor_dim0_stride @ bits 191:160
        g1[6] = 0;
        g1[7] = 0;
#if __clang_major__ >= 23
        i32x8 g4 = {0, 0, 0, 0, 0, 0, 0, 0};
        __builtin_amdgcn_tensor_load_to_lds(g0, g1, g2, g3, g4, 0);
#else
        __builtin_amdgcn_tensor_load_to_lds(g0, g1, g2, g3, 0);
#endif
        __builtin_amdgcn_s_wait_tensorcnt(0);
    }
    asm volatile("" : : "r"(&Asf[0]) : "memory");  // TDM wrote LDS
    __syncthreads();
#else
    for (int i = tid; i < 16 * 256; i += 256) {
        int r = i >> 8, c = i & 255;
        Asf[r * 260 + c] = A[(size_t)(m0 + r) * EMBED + c];
    }
    __syncthreads();
#endif

    const int lane = tid & 31;
    const int wave = tid >> 5;
    const int row  = lane & 15;
    const int kg   = lane >> 4;
    const int ntiles = N >> 4;

    for (int nt = wave; nt < ntiles; nt += 8) {
        v8f acc = {0.f, 0.f, 0.f, 0.f, 0.f, 0.f, 0.f, 0.f};
        const _Float16* bp = Bp + (size_t)nt * 8192 + lane * 32;
#pragma unroll
        for (int ks = 0; ks < 8; ++ks) {
            // A fragment: two contiguous 8-float runs per lane, split hi/lo
            const float* ap = &Asf[row * 260 + ks * 32 + (kg << 3)];
            v16h ah, al;
#pragma unroll
            for (int e = 0; e < 8; ++e) {
                float f0 = ap[e], f1 = ap[e + 16];
                _Float16 h0 = (_Float16)f0, h1 = (_Float16)f1;
                ah[e]     = h0; al[e]     = (_Float16)(f0 - (float)h0);
                ah[e + 8] = h1; al[e + 8] = (_Float16)(f1 - (float)h1);
            }
            v16h bh = *(const v16h*)(bp + (size_t)ks * 1024);
            v16h bl = *(const v16h*)(bp + (size_t)ks * 1024 + 16);
            acc = __builtin_amdgcn_wmma_f32_16x16x32_f16(
                false, ah, false, bh, (short)0, acc, false, false);
            acc = __builtin_amdgcn_wmma_f32_16x16x32_f16(
                false, al, false, bh, (short)0, acc, false, false);
            acc = __builtin_amdgcn_wmma_f32_16x16x32_f16(
                false, ah, false, bl, (short)0, acc, false, false);
        }
        const int col = nt * 16 + (lane & 15);
        const float bv = bias[col];
#pragma unroll
        for (int g = 0; g < 8; ++g) {
            int r = m0 + g + (kg << 3);
            C[(size_t)r * N + col] = acc[g] + bv;
        }
    }
}

// ---------------------------------------------------------------------------
// Deformable sampling. One wave per (b, q, head); lane = channel (d=32).
// Softmax over the 8 logits fused in. A prefetch pass issues
// global_prefetch_b8 for all 32 corner cachelines (vproj is L2-resident:
// 118MB < 192MB), then the weighted bilinear gather runs with latency hidden.
// ---------------------------------------------------------------------------
__global__ __launch_bounds__(256)
void msda_sample_kernel(const float* __restrict__ vproj,
                        const float* __restrict__ so,
                        const float* __restrict__ attn,
                        const float* __restrict__ ref,
                        const int*   __restrict__ ss,
                        float* __restrict__ out,
                        int nq, int nv) {
    const int lane = threadIdx.x & 31;
    const int wave = threadIdx.x >> 5;
    const int flat = blockIdx.x * 8 + wave;    // (b*nq + q)*HEADS + h
    const int h  = flat & (HEADS - 1);
    const int bq = flat >> 3;                  // b*nq + q
    const int b  = bq / nq;

    const int Hh = ss[0];
    const int Ww = ss[1];
    const float fH = (float)Hh, fW = (float)Ww;

    // fused softmax over 8 attention logits
    const float* lg = attn + (size_t)bq * (HEADS * NPTS) + h * NPTS;
    float l[NPTS];
    float mx = -3.402823466e+38f;
#pragma unroll
    for (int p = 0; p < NPTS; ++p) { l[p] = lg[p]; mx = fmaxf(mx, l[p]); }
    float s = 0.f;
#pragma unroll
    for (int p = 0; p < NPTS; ++p) { l[p] = __expf(l[p] - mx); s += l[p]; }
    const float inv = 1.f / s;

    const float* sop = so  + (size_t)bq * (HEADS * NPTS * 2) + h * (NPTS * 2);
    const float* rp  = ref + (size_t)bq * 8;   // (NUM_Z=4, 2)
    const float* vbase = vproj + (size_t)b * nv * EMBED + h * DHEAD + lane;

    // compute all sample coordinates once
    float xs[NPTS], ys[NPTS];
#pragma unroll
    for (int p = 0; p < NPTS; ++p) {
        const int z = p & 3;   // reshape (..,2,4,2): ref z index = p % 4
        xs[p] = rp[z * 2 + 0] * fW + sop[p * 2 + 0] - 0.5f;
        ys[p] = rp[z * 2 + 1] * fH + sop[p * 2 + 1] - 0.5f;
    }

    // prefetch pass: all 4 corners of all 8 points
#pragma unroll
    for (int p = 0; p < NPTS; ++p) {
        int ix0 = (int)floorf(xs[p]);
        int iy0 = (int)floorf(ys[p]);
#pragma unroll
        for (int cy = 0; cy < 2; ++cy) {
#pragma unroll
            for (int cx = 0; cx < 2; ++cx) {
                int cxi = min(max(ix0 + cx, 0), Ww - 1);
                int cyi = min(max(iy0 + cy, 0), Hh - 1);
                __builtin_prefetch(vbase + (size_t)(cyi * Ww + cxi) * EMBED, 0, 1);
            }
        }
    }

    float acc = 0.f;
#pragma unroll
    for (int p = 0; p < NPTS; ++p) {
        float x = xs[p], y = ys[p];
        float x0f = floorf(x), y0f = floorf(y);
        int ix0 = (int)x0f, iy0 = (int)y0f;
        float wx1 = x - x0f, wy1 = y - y0f;
        float wx0 = 1.f - wx1, wy0 = 1.f - wy1;

        float pacc = 0.f;
#pragma unroll
        for (int cy = 0; cy < 2; ++cy) {
#pragma unroll
            for (int cx = 0; cx < 2; ++cx) {
                int ix = ix0 + cx, iy = iy0 + cy;
                bool valid = (ix >= 0) & (ix < Ww) & (iy >= 0) & (iy < Hh);
                int cxi = min(max(ix, 0), Ww - 1);
                int cyi = min(max(iy, 0), Hh - 1);
                float v = vbase[(size_t)(cyi * Ww + cxi) * EMBED];
                float w = (cx ? wx1 : wx0) * (cy ? wy1 : wy0);
                pacc += valid ? (w * v) : 0.f;
            }
        }
        acc += (l[p] * inv) * pacc;
    }
    out[(size_t)bq * EMBED + h * DHEAD + lane] = acc;
}

// ---------------------------------------------------------------------------
// Host launcher.
// Inputs: 0 query (bs,nq,256) 1 value (bs,nv,256) 2 reference_points
// (bs,nq,4,2) 3 W_off (256,128) 4 b_off 5 W_attn (256,64) 6 b_attn
// 7 W_val (256,256) 8 b_val 9 spatial_shapes (1,2) i32
// ---------------------------------------------------------------------------
extern "C" void kernel_launch(void* const* d_in, const int* in_sizes, int n_in,
                              void* d_out, int out_size, void* d_ws, size_t ws_size,
                              hipStream_t stream) {
    const float* query  = (const float*)d_in[0];
    const float* value  = (const float*)d_in[1];
    const float* ref    = (const float*)d_in[2];
    const float* W_off  = (const float*)d_in[3];
    const float* b_off  = (const float*)d_in[4];
    const float* W_attn = (const float*)d_in[5];
    const float* b_attn = (const float*)d_in[6];
    const float* W_val  = (const float*)d_in[7];
    const float* b_val  = (const float*)d_in[8];
    const int*   ss     = (const int*)d_in[9];
    float* out = (float*)d_out;

    const int bs = 2;
    const int nq = in_sizes[0] / (bs * EMBED);   // 4096
    const int nv = in_sizes[1] / (bs * EMBED);   // 57600
    const int Mv = bs * nv;                      // 115200 (16-divisible)
    const int Mq = bs * nq;                      // 8192

    // workspace layout (float units); packed weights hold hi+lo halves
    float* ws = (float*)d_ws;
    _Float16* pWval  = (_Float16*)(ws);            // 512*256 halves = 64K f
    _Float16* pWoff  = (_Float16*)(ws + 65536);    // 512*128 halves = 32K f
    _Float16* pWattn = (_Float16*)(ws + 98304);    // 512*64  halves = 16K f
    float* vproj   = ws + 131072;                  // bs*nv*256
    float* sobuf   = vproj + (size_t)Mv * EMBED;   // bs*nq*128
    float* attnbuf = sobuf + (size_t)Mq * 128;     // bs*nq*64

    // pack weights into split-f16 WMMA fragment order (tiny, once per launch)
    pack_b_kernel<<<(EMBED * 256 * 2 + 255) / 256, 256, 0, stream>>>(W_val,  pWval,  256);
    pack_b_kernel<<<(EMBED * 128 * 2 + 255) / 256, 256, 0, stream>>>(W_off,  pWoff,  128);
    pack_b_kernel<<<(EMBED *  64 * 2 + 255) / 256, 256, 0, stream>>>(W_attn, pWattn,  64);

    // value projection: v = value @ W_val + b_val   (the big streaming GEMM)
    gemm_wmma_kernel<<<Mv / 16, 256, 0, stream>>>(value, pWval, b_val, vproj, 256);
    // sampling offsets: so = query @ W_off + b_off
    gemm_wmma_kernel<<<Mq / 16, 256, 0, stream>>>(query, pWoff, b_off, sobuf, 128);
    // attention logits: query @ W_attn + b_attn (softmax fused into sampler)
    gemm_wmma_kernel<<<Mq / 16, 256, 0, stream>>>(query, pWattn, b_attn, attnbuf, 64);

    // deformable bilinear sampling + weighted reduce
    msda_sample_kernel<<<(Mq * HEADS) / 8, 256, 0, stream>>>(
        vproj, sobuf, attnbuf, ref, ss, out, nq, nv);
}
